// ActorCriticLSTM_81887846465659
// MI455X (gfx1250) — compile-verified, matrix operand order checked by reading, not run on hardware
//
#include <hip/hip_runtime.h>
#include <hip/hip_bf16.h>
#include <math.h>

// ---------------------------------------------------------------------------
// Types for CDNA5 WMMA (wave32): D(16x16 f32) = A(16x32 f16) x B(32x16 f16) + C
// ---------------------------------------------------------------------------
typedef __attribute__((ext_vector_type(16))) _Float16 v16h;
typedef __attribute__((ext_vector_type(8)))  _Float16 v8h;
typedef __attribute__((ext_vector_type(8)))  float    v8f;
typedef __attribute__((ext_vector_type(4)))  float    v4f;

#define WMMA16(a, b, c) \
  __builtin_amdgcn_wmma_f32_16x16x32_f16(false, (a), false, (b), (short)0, (c), false, false)

#define SEQ   80
#define BATCH 128
#define NTOT  (SEQ * BATCH)     // 10240
#define HID   256
#define FLAT  2592              // 32*9*9
#define NOUT  18

// A/B fragment loader for a row-major f16 matrix with row stride that is a
// multiple of 8 elements.  Lane L supplies row (L&15); khalf = L>>4 selects the
// K-halves per the 16-bit A/B VGPR layout:
//   elems 0..7  -> K = kb*32 + khalf*8 + {0..7}
//   elems 8..15 -> K = kb*32 + 16 + khalf*8 + {0..7}
static __device__ __forceinline__ v16h frag_ld(const _Float16* __restrict__ row,
                                               int kb, int khalf) {
  v8h lo = *(const v8h*)(row + kb * 32 + khalf * 8);
  v8h hi = *(const v8h*)(row + kb * 32 + 16 + khalf * 8);
  v16h r;
#pragma unroll
  for (int j = 0; j < 8; ++j) { r[j] = lo[j]; r[8 + j] = hi[j]; }
  return r;
}

// 4 f16 from a 4-byte-aligned pointer into fragment elements e0..e0+3.
static __device__ __forceinline__ void ld4h(const _Float16* __restrict__ p,
                                            v16h& a, int e0) {
  const unsigned* q = (const unsigned*)p;
  unsigned u0 = q[0], u1 = q[1];
  union { unsigned u; _Float16 h[2]; } c0, c1;
  c0.u = u0; c1.u = u1;
  a[e0 + 0] = c0.h[0]; a[e0 + 1] = c0.h[1];
  a[e0 + 2] = c1.h[0]; a[e0 + 3] = c1.h[1];
}

// ---------------------------------------------------------------------------
// f32 -> f16 conversion (weights)
// ---------------------------------------------------------------------------
__global__ void cvt_f16_kernel(const float* __restrict__ in,
                               _Float16* __restrict__ out, int n) {
  int i = blockIdx.x * 256 + threadIdx.x;
  if (i < n) out[i] = (_Float16)in[i];
}

// ---------------------------------------------------------------------------
// init LSTM state: h -> f16 ping buffer, c -> f32 buffer
// ---------------------------------------------------------------------------
__global__ void init_state_kernel(const float* __restrict__ h0,
                                  const float* __restrict__ c0,
                                  _Float16* __restrict__ h16,
                                  float* __restrict__ cbuf) {
  int i = blockIdx.x * 256 + threadIdx.x;
  if (i < BATCH * HID) { h16[i] = (_Float16)h0[i]; cbuf[i] = c0[i]; }
}

// ---------------------------------------------------------------------------
// conv1: 4->16 ch, 8x8, stride 4, 84x84 -> 20x20, implicit GEMM K=256
// one wave = 16 positions x 16 channels; 25 position tiles per sample
// ---------------------------------------------------------------------------
__global__ __launch_bounds__(128) void conv1_kernel(
    const float* __restrict__ obs, const _Float16* __restrict__ w1,
    const float* __restrict__ b1, _Float16* __restrict__ y1) {
  int wave = threadIdx.x >> 5, lane = threadIdx.x & 31;
  long gw = (long)blockIdx.x * 4 + wave;            // 10240*25 waves
  long n = gw / 25; int ptile = (int)(gw % 25);
  int mrow = lane & 15, khalf = lane >> 4;

  int pos = ptile * 16 + mrow;
  int oy = pos / 20, ox = pos % 20;
  const float* base = obs + n * (4 * 84 * 84);
  const _Float16* wrow = w1 + mrow * 256;           // lane column = out channel

  v8f acc; float bias = b1[mrow];
#pragma unroll
  for (int i = 0; i < 8; ++i) acc[i] = bias;

#pragma unroll
  for (int kb = 0; kb < 8; ++kb) {
    int c = kb >> 1;
    int kh0 = (kb & 1) * 4 + khalf;                 // rows kh0 and kh0+2
    const float* p0 = base + c * 7056 + (oy * 4 + kh0) * 84 + ox * 4;
    const float* p1 = p0 + 2 * 84;
    v4f f0 = *(const v4f*)p0;
    v4f f1 = *(const v4f*)(p0 + 4);
    v4f f2 = *(const v4f*)p1;
    v4f f3 = *(const v4f*)(p1 + 4);
    v16h a;
#pragma unroll
    for (int j = 0; j < 4; ++j) {
      a[j]      = (_Float16)f0[j];  a[4 + j]  = (_Float16)f1[j];
      a[8 + j]  = (_Float16)f2[j];  a[12 + j] = (_Float16)f3[j];
    }
    v16h b = frag_ld(wrow, kb, khalf);
    acc = WMMA16(a, b, acc);
  }

  int oc = lane & 15;
  int ps = ptile * 16 + ((lane >> 4) * 8);          // D: M = vgpr + (hi?8)
  v8h o;
#pragma unroll
  for (int v = 0; v < 8; ++v) o[v] = (_Float16)fmaxf(acc[v], 0.0f);
  *(v8h*)(y1 + (n * 16 + oc) * 400 + ps) = o;       // ps mult of 8 -> aligned
}

// ---------------------------------------------------------------------------
// conv2: 16->32 ch, 4x4, stride 2, 20x20 -> 9x9, implicit GEMM K=256
// one wave = 16 positions x 32 channels (two N tiles); 6 pos tiles per sample
// output in FC flat order: x2[n][oc*81+pos]
// ---------------------------------------------------------------------------
__global__ __launch_bounds__(128) void conv2_kernel(
    const _Float16* __restrict__ y1, const _Float16* __restrict__ w2,
    const float* __restrict__ b2, _Float16* __restrict__ x2) {
  int wave = threadIdx.x >> 5, lane = threadIdx.x & 31;
  long gw = (long)blockIdx.x * 4 + wave;            // 10240*6 waves
  long n = gw / 6; int ptile = (int)(gw % 6);
  int mrow = lane & 15, khalf = lane >> 4;

  int pos = ptile * 16 + mrow; if (pos > 80) pos = 80;   // clamp loads
  int oy = pos / 9, ox = pos % 9;
  const _Float16* base = y1 + n * 6400;
  const _Float16* w2a = w2 + mrow * 256;
  const _Float16* w2b = w2 + (16 + mrow) * 256;

  v8f acc0, acc1;
  float bi0 = b2[mrow], bi1 = b2[16 + mrow];
#pragma unroll
  for (int i = 0; i < 8; ++i) { acc0[i] = bi0; acc1[i] = bi1; }

#pragma unroll
  for (int kb = 0; kb < 8; ++kb) {
    int ic0 = kb * 2;
    int iy = oy * 2 + khalf * 2, ix = ox * 2;
    const _Float16* p00 = base + ic0 * 400 + iy * 20 + ix;   // ic0, kh=2*khalf
    const _Float16* p01 = p00 + 20;                          // ic0, kh=2*khalf+1
    const _Float16* p10 = p00 + 400;                         // ic0+1
    const _Float16* p11 = p01 + 400;
    v16h a;
    ld4h(p00, a, 0); ld4h(p01, a, 4); ld4h(p10, a, 8); ld4h(p11, a, 12);
    v16h bA = frag_ld(w2a, kb, khalf);
    v16h bB = frag_ld(w2b, kb, khalf);
    acc0 = WMMA16(a, bA, acc0);
    acc1 = WMMA16(a, bB, acc1);
  }

  int oc = lane & 15;
  int ps = ptile * 16 + ((lane >> 4) * 8);
  _Float16* row = x2 + n * 2592;
#pragma unroll
  for (int v = 0; v < 8; ++v) {
    int p = ps + v;
    if (p < 81) {
      row[oc * 81 + p]        = (_Float16)fmaxf(acc0[v], 0.0f);
      row[(oc + 16) * 81 + p] = (_Float16)fmaxf(acc1[v], 0.0f);
    }
  }
}

// ---------------------------------------------------------------------------
// FC: (10240 x 2592) @ fc_w^T (2592 x 256) + b, ReLU -> x3 f16
// ---------------------------------------------------------------------------
__global__ __launch_bounds__(128) void fc_kernel(
    const _Float16* __restrict__ x2, const _Float16* __restrict__ fcw,
    const float* __restrict__ fcb, _Float16* __restrict__ x3) {
  int wave = threadIdx.x >> 5, lane = threadIdx.x & 31;
  long gw = (long)blockIdx.x * 4 + wave;            // 640*16 waves
  int mtile = (int)(gw >> 4), ntile = (int)(gw & 15);
  int mrow = lane & 15, khalf = lane >> 4;

  const _Float16* arow = x2 + (long)(mtile * 16 + mrow) * 2592;
  const _Float16* brow = fcw + (long)(ntile * 16 + mrow) * 2592;

  v8f acc; float bias = fcb[ntile * 16 + (lane & 15)];
#pragma unroll
  for (int i = 0; i < 8; ++i) acc[i] = bias;

  for (int kb = 0; kb < 81; ++kb) {
    v16h a = frag_ld(arow, kb, khalf);
    v16h b = frag_ld(brow, kb, khalf);
    acc = WMMA16(a, b, acc);
  }

  int ncol = ntile * 16 + (lane & 15);
  int m0 = mtile * 16 + ((lane >> 4) * 8);
#pragma unroll
  for (int v = 0; v < 8; ++v)
    x3[(long)(m0 + v) * 256 + ncol] = (_Float16)fmaxf(acc[v], 0.0f);
}

// ---------------------------------------------------------------------------
// xgates: (10240 x 256) @ w_ih^T (256 x 1024) + b_ih + b_hh -> f32
// ---------------------------------------------------------------------------
__global__ __launch_bounds__(128) void xgates_kernel(
    const _Float16* __restrict__ x3, const _Float16* __restrict__ wih,
    const float* __restrict__ bih, const float* __restrict__ bhh,
    float* __restrict__ xg) {
  int wave = threadIdx.x >> 5, lane = threadIdx.x & 31;
  long gw = (long)blockIdx.x * 4 + wave;            // 640*64 waves
  int mtile = (int)(gw / 64), ntile = (int)(gw % 64);
  int mrow = lane & 15, khalf = lane >> 4;

  const _Float16* arow = x3 + (long)(mtile * 16 + mrow) * 256;
  const _Float16* brow = wih + (long)(ntile * 16 + mrow) * 256;

  int ncol = ntile * 16 + (lane & 15);
  v8f acc; float bias = bih[ncol] + bhh[ncol];
#pragma unroll
  for (int i = 0; i < 8; ++i) acc[i] = bias;

#pragma unroll
  for (int kb = 0; kb < 8; ++kb) {
    v16h a = frag_ld(arow, kb, khalf);
    v16h b = frag_ld(brow, kb, khalf);
    acc = WMMA16(a, b, acc);
  }

  int m0 = mtile * 16 + ((lane >> 4) * 8);
#pragma unroll
  for (int v = 0; v < 8; ++v)
    xg[(long)(m0 + v) * 1024 + ncol] = acc[v];
}

// ---------------------------------------------------------------------------
// one LSTM step: gates = xg[t] + h_prev @ w_hh^T ; cell ; mask on carry only
// block = 4 waves, one gate tile each; grid = (128/16)*(256/16) = 128 blocks
// ---------------------------------------------------------------------------
__global__ __launch_bounds__(128) void lstm_step_kernel(
    int t, const _Float16* __restrict__ hprev, _Float16* __restrict__ hnext,
    float* __restrict__ cbuf, const float* __restrict__ xg,
    const _Float16* __restrict__ whh, const float* __restrict__ mask,
    float* __restrict__ hs) {
  __shared__ float gbuf[4][16][16];
  int gate = threadIdx.x >> 5, lane = threadIdx.x & 31;
  int btile = blockIdx.x >> 4, htile = blockIdx.x & 15;
  int mrow = lane & 15, khalf = lane >> 4;

  int colbase = gate * 256 + htile * 16;
  const _Float16* arow = hprev + (btile * 16 + mrow) * 256;
  const _Float16* brow = whh + (long)(colbase + mrow) * 256;

  int ncol = colbase + (lane & 15);
  int hi8 = (lane >> 4) * 8;
  int m0 = btile * 16 + hi8;

  v8f acc;
#pragma unroll
  for (int v = 0; v < 8; ++v)
    acc[v] = xg[((long)t * 128 + m0 + v) * 1024 + ncol];

#pragma unroll
  for (int kb = 0; kb < 8; ++kb) {
    v16h a = frag_ld(arow, kb, khalf);
    v16h b = frag_ld(brow, kb, khalf);
    acc = WMMA16(a, b, acc);
  }

#pragma unroll
  for (int v = 0; v < 8; ++v) gbuf[gate][v + hi8][lane & 15] = acc[v];
  __syncthreads();

  for (int e = threadIdx.x; e < 256; e += 128) {
    int m = e >> 4, nn = e & 15;
    int b = btile * 16 + m, hid = htile * 16 + nn;
    float iv = gbuf[0][m][nn], fv = gbuf[1][m][nn];
    float gv = gbuf[2][m][nn], ov = gbuf[3][m][nn];
    float cp = cbuf[b * 256 + hid];
    float ig = 1.0f / (1.0f + expf(-iv));
    float fg = 1.0f / (1.0f + expf(-fv));
    float og = 1.0f / (1.0f + expf(-ov));
    float cn = fg * cp + ig * tanhf(gv);
    float hn = og * tanhf(cn);
    hs[((long)t * 128 + b) * 256 + hid] = hn;        // unmasked output
    float mk = mask[t * 128 + b];                    // mask the carry
    hnext[b * 256 + hid] = (_Float16)(hn * mk);
    cbuf[b * 256 + hid] = cn * mk;
  }
}

// ---------------------------------------------------------------------------
// heads: value, log-softmax(logits), take-along-axis, entropy
// ---------------------------------------------------------------------------
__global__ __launch_bounds__(256) void heads_kernel(
    const float* __restrict__ hs, const float* __restrict__ vw,
    const float* __restrict__ vb, const float* __restrict__ lw,
    const float* __restrict__ lb, const int* __restrict__ acts,
    float* __restrict__ out) {
  __shared__ float slw[NOUT * HID];
  __shared__ float svw[HID];
  __shared__ float slb[NOUT];
  for (int i = threadIdx.x; i < NOUT * HID; i += 256) slw[i] = lw[i];
  if (threadIdx.x < HID) svw[threadIdx.x] = vw[threadIdx.x];
  if (threadIdx.x < NOUT) slb[threadIdx.x] = lb[threadIdx.x];
  __syncthreads();

  int r = blockIdx.x * 256 + threadIdx.x;            // r < 10240
  const float* h = hs + (long)r * HID;
  float lg[NOUT]; float val = vb[0];
#pragma unroll
  for (int j = 0; j < NOUT; ++j) lg[j] = slb[j];
  for (int k = 0; k < HID; ++k) {
    float hv = h[k];
    val += hv * svw[k];
#pragma unroll
    for (int j = 0; j < NOUT; ++j) lg[j] += hv * slw[j * HID + k];
  }
  float mx = lg[0];
#pragma unroll
  for (int j = 1; j < NOUT; ++j) mx = fmaxf(mx, lg[j]);
  float se = 0.0f;
#pragma unroll
  for (int j = 0; j < NOUT; ++j) se += expf(lg[j] - mx);
  float lse = mx + logf(se);
  float ent = 0.0f;
#pragma unroll
  for (int j = 0; j < NOUT; ++j) {
    float lp = lg[j] - lse;
    ent -= expf(lp) * lp;
  }
  int a = acts[r];
  out[r]             = lg[a] - lse;
  out[NTOT + r]      = ent;
  out[2 * NTOT + r]  = val;
}

// ---------------------------------------------------------------------------
// host launcher
// ---------------------------------------------------------------------------
extern "C" void kernel_launch(void* const* d_in, const int* in_sizes, int n_in,
                              void* d_out, int out_size, void* d_ws, size_t ws_size,
                              hipStream_t stream) {
  const float* obs   = (const float*)d_in[0];
  const float* mask  = (const float*)d_in[1];
  const float* h0    = (const float*)d_in[2];
  const float* c0    = (const float*)d_in[3];
  const int*   acts  = (const int*)d_in[4];
  const float* c1w   = (const float*)d_in[5];
  const float* c1b   = (const float*)d_in[6];
  const float* c2w   = (const float*)d_in[7];
  const float* c2b   = (const float*)d_in[8];
  const float* fcw   = (const float*)d_in[9];
  const float* fcb   = (const float*)d_in[10];
  const float* wih   = (const float*)d_in[11];
  const float* whh   = (const float*)d_in[12];
  const float* bih   = (const float*)d_in[13];
  const float* bhh   = (const float*)d_in[14];
  const float* vw    = (const float*)d_in[15];
  const float* vb    = (const float*)d_in[16];
  const float* lw    = (const float*)d_in[17];
  const float* lb    = (const float*)d_in[18];
  float* out = (float*)d_out;

  // workspace carve-up (256B aligned)
  char* ws = (char*)d_ws;
  size_t off = 0;
  auto take = [&](size_t bytes) {
    char* p = ws + off;
    off += (bytes + 255) & ~(size_t)255;
    return p;
  };
  _Float16* w1h  = (_Float16*)take(4096 * 2);
  _Float16* w2h  = (_Float16*)take(8192 * 2);
  _Float16* fcwh = (_Float16*)take((size_t)HID * FLAT * 2);
  _Float16* wihh = (_Float16*)take((size_t)1024 * HID * 2);
  _Float16* whhh = (_Float16*)take((size_t)1024 * HID * 2);
  _Float16* y1   = (_Float16*)take((size_t)NTOT * 6400 * 2);
  _Float16* x2   = (_Float16*)take((size_t)NTOT * FLAT * 2);
  _Float16* x3   = (_Float16*)take((size_t)NTOT * HID * 2);
  float*    xg   = (float*)   take((size_t)NTOT * 1024 * 4);
  float*    hsb  = (float*)   take((size_t)NTOT * HID * 4);
  _Float16* hA   = (_Float16*)take((size_t)BATCH * HID * 2);
  _Float16* hB   = (_Float16*)take((size_t)BATCH * HID * 2);
  float*    cb   = (float*)   take((size_t)BATCH * HID * 4);
  (void)ws_size; (void)n_in; (void)in_sizes; (void)out_size;

  // weight conversions
  cvt_f16_kernel<<<(4096 + 255) / 256, 256, 0, stream>>>(c1w, w1h, 4096);
  cvt_f16_kernel<<<(8192 + 255) / 256, 256, 0, stream>>>(c2w, w2h, 8192);
  cvt_f16_kernel<<<(HID * FLAT + 255) / 256, 256, 0, stream>>>(fcw, fcwh, HID * FLAT);
  cvt_f16_kernel<<<(1024 * HID + 255) / 256, 256, 0, stream>>>(wih, wihh, 1024 * HID);
  cvt_f16_kernel<<<(1024 * HID + 255) / 256, 256, 0, stream>>>(whh, whhh, 1024 * HID);
  init_state_kernel<<<(BATCH * HID + 255) / 256, 256, 0, stream>>>(h0, c0, hA, cb);

  // trunk
  conv1_kernel<<<(NTOT * 25) / 4, 128, 0, stream>>>(obs, w1h, c1b, y1);
  conv2_kernel<<<(NTOT * 6) / 4, 128, 0, stream>>>(y1, w2h, c2b, x2);
  fc_kernel<<<(640 * 16) / 4, 128, 0, stream>>>(x2, fcwh, fcb, x3);
  xgates_kernel<<<(640 * 64) / 4, 128, 0, stream>>>(x3, wihh, bih, bhh, xg);

  // sequential LSTM scan (stream order carries the h/c dependency)
  for (int t = 0; t < SEQ; ++t) {
    const _Float16* hp = (t & 1) ? hB : hA;
    _Float16*       hn = (t & 1) ? hA : hB;
    lstm_step_kernel<<<128, 128, 0, stream>>>(t, hp, hn, cb, xg, whhh, mask, hsb);
  }

  // heads
  heads_kernel<<<NTOT / 256, 256, 0, stream>>>(hsb, vw, vb, lw, lb, acts, out);
}
